// QuantumSelfAttention_91199335563801
// MI455X (gfx1250) — matrix operand order repaired and checked by reading, not compiled, not run on hardware
//
#include <hip/hip_runtime.h>
#include <hip/hip_bf16.h>
#include <math.h>

#define B_  8
#define S_  1024
#define DK_ 64
#define NQ_ 8

typedef __attribute__((ext_vector_type(16))) __bf16 v16bf;
typedef __attribute__((ext_vector_type(8)))  float  v8f;

// ---------------------------------------------------------------------------
// Kernel 1: per-token single-qubit states for Q and K.
// amp0 = A0*c - B0*s ; amp1 = conj(B0)*c + conj(A0)*s
// with A0 = e^{-i(phi+om)/2} cos(th/2), B0 = e^{+i(phi-om)/2} sin(th/2),
// c = cos(tanh(x)*pi/2), s = sin(tanh(x)*pi/2).
// Layout: st[(b*S + s)*NQ + n] = {amp0.re, amp0.im, amp1.re, amp1.im}
// ---------------------------------------------------------------------------
__device__ __forceinline__ float4 qubit_state(const float* th, int n, float c, float s) {
  float phi = th[n * 3 + 0], tt = th[n * 3 + 1], om = th[n * 3 + 2];
  float ct = cosf(0.5f * tt), st = sinf(0.5f * tt);
  float csh, ssh, cdh, sdh;
  sincosf(0.5f * (phi + om), &ssh, &csh);
  sincosf(0.5f * (phi - om), &sdh, &cdh);
  float A0r = csh * ct, A0i = -ssh * ct;   // e_mpo * ct
  float B0r = cdh * st, B0i =  sdh * st;   // e_pmo * st
  float4 v;
  v.x = A0r * c - B0r * s;
  v.y = A0i * c - B0i * s;
  v.z = B0r * c + A0r * s;
  v.w = -(B0i * c + A0i * s);
  return v;
}

__global__ void qsa_states(const float* __restrict__ x,
                           const float* __restrict__ thq,
                           const float* __restrict__ thk,
                           float4* __restrict__ qst,
                           float4* __restrict__ kst) {
  int idx = blockIdx.x * blockDim.x + threadIdx.x;   // b*S*NQ + s*NQ + n
  if (idx >= B_ * S_ * NQ_) return;
  int n  = idx & (NQ_ - 1);
  int bs = idx >> 3;
  float xn = tanhf(x[(size_t)bs * DK_ + n]);
  float c, s;
  sincosf(xn * 1.57079632679489662f, &s, &c);        // angle = xn*pi, halved
  qst[idx] = qubit_state(thq, n, c, s);
  kst[idx] = qubit_state(thk, n, c, s);
}

// ---------------------------------------------------------------------------
// Kernel 2: scores + softmax for one row (b,s) per 256-thread block.
// score(s,t) = (prod_n |<q_sn|k_tn>|^2 + 1) * 0.5 ; softmax over t.
// ---------------------------------------------------------------------------
__global__ void __launch_bounds__(256)
qsa_attn(const float4* __restrict__ qst,
         const float4* __restrict__ kst,
         float* __restrict__ attn) {
  __shared__ float4 qsh[NQ_];
  __shared__ float  red[8];
  __shared__ float  bcast;
  int tid = threadIdx.x;
  int row = blockIdx.x;                 // b*S + s
  int b   = row >> 10;
  if (tid < NQ_) qsh[tid] = qst[(size_t)row * NQ_ + tid];
  __syncthreads();

  float sc[4];
  #pragma unroll
  for (int i = 0; i < 4; ++i) {
    int t = tid + i * 256;
    const float4* kp = kst + ((size_t)b * S_ + t) * NQ_;
    float core = 1.0f;
    #pragma unroll
    for (int n = 0; n < NQ_; ++n) {
      float4 k4 = kp[n];
      float4 q4 = qsh[n];
      float re = q4.x * k4.x + q4.y * k4.y + q4.z * k4.z + q4.w * k4.w;
      float im = q4.x * k4.y - q4.y * k4.x + q4.z * k4.w - q4.w * k4.z;
      core *= re * re + im * im;
    }
    sc[i] = (core + 1.0f) * 0.5f;
  }

  // block max
  float m = fmaxf(fmaxf(sc[0], sc[1]), fmaxf(sc[2], sc[3]));
  for (int off = 16; off; off >>= 1) m = fmaxf(m, __shfl_xor(m, off, 32));
  if ((tid & 31) == 0) red[tid >> 5] = m;
  __syncthreads();
  if (tid == 0) {
    float mm = red[0];
    for (int i = 1; i < 8; ++i) mm = fmaxf(mm, red[i]);
    bcast = mm;
  }
  __syncthreads();
  float rowmax = bcast;

  // exp + block sum
  float e[4], ssum = 0.0f;
  #pragma unroll
  for (int i = 0; i < 4; ++i) { e[i] = expf(sc[i] - rowmax); ssum += e[i]; }
  for (int off = 16; off; off >>= 1) ssum += __shfl_xor(ssum, off, 32);
  if ((tid & 31) == 0) red[tid >> 5] = ssum;
  __syncthreads();
  if (tid == 0) {
    float t = 0.0f;
    for (int i = 0; i < 8; ++i) t += red[i];
    bcast = t;
  }
  __syncthreads();
  float inv = 1.0f / bcast;

  float* arow = attn + (size_t)row * S_;
  #pragma unroll
  for (int i = 0; i < 4; ++i) arow[tid + i * 256] = e[i] * inv;
}

// ---------------------------------------------------------------------------
// Kernel 3: values = x @ W_v^T + b_v, stored TRANSPOSED in bf16:
// vT[(b*DK + d)*S + t] so WMMA B-fragments are K-contiguous.
// ---------------------------------------------------------------------------
__global__ void qsa_values(const float* __restrict__ x,
                           const float* __restrict__ Wv,
                           const float* __restrict__ bv,
                           __hip_bfloat16* __restrict__ vT) {
  __shared__ float xs[DK_];
  int row = blockIdx.x;                 // b*S + t
  int d   = threadIdx.x;                // 0..63
  xs[d] = x[(size_t)row * DK_ + d];
  __syncthreads();
  const float* w = Wv + (size_t)d * DK_;
  float acc = bv[d];
  #pragma unroll 8
  for (int k = 0; k < DK_; ++k) acc = fmaf(xs[k], w[k], acc);
  int b = row >> 10, t = row & (S_ - 1);
  vT[((size_t)(b * DK_ + d)) * S_ + t] = __float2bfloat16(acc);
}

// ---------------------------------------------------------------------------
// Kernel 4: out = attn @ values via v_wmma_f32_16x16x32_bf16.
// One wave per 16x16 output tile; K-loop in steps of 32.
// A (16x32 bf16) lane layout: row M = lane%16, half h = lane/16,
//   elements 0..15 -> K = 16h+0..15
//   (four contiguous 16B f32 loads from the attn row, cvt to bf16).
// B (32x16 bf16): col N = lane%16, elements 0..15 -> K = 16h+0..15
//   (one 32B contiguous bf16 vector load from vT).
// D (16x16 f32): VGPR r -> row r+8h, col lane%16.
// ---------------------------------------------------------------------------
__global__ void __launch_bounds__(256)
qsa_out_wmma(const float* __restrict__ attn,
             const __hip_bfloat16* __restrict__ vT_,
             float* __restrict__ out) {
  const __bf16* vT = (const __bf16*)vT_;
  int lane = threadIdx.x & 31;
  int wave = threadIdx.x >> 5;
  int gw   = blockIdx.x * 8 + wave;     // 0..2047 tiles total
  int b    = gw >> 8;                   // 256 tiles per batch
  int rem  = gw & 255;
  int sTile = rem >> 2;                 // 64 s-tiles
  int dTile = rem & 3;                  // 4 d-tiles
  int m = lane & 15;
  int h = lane >> 4;

  const float*  arow = attn + ((size_t)(b * S_)  + sTile * 16 + m) * S_;
  const __bf16* brow = vT   + ((size_t)(b * DK_) + dTile * 16 + m) * S_;

  v8f acc = {};
  for (int k0 = 0; k0 < S_; k0 += 32) {
    const float4 a0 = *(const float4*)(arow + k0 + 16 * h);
    const float4 a1 = *(const float4*)(arow + k0 + 16 * h + 4);
    const float4 a2 = *(const float4*)(arow + k0 + 16 * h + 8);
    const float4 a3 = *(const float4*)(arow + k0 + 16 * h + 12);
    v16bf A;
    A[0]  = (__bf16)a0.x; A[1]  = (__bf16)a0.y; A[2]  = (__bf16)a0.z; A[3]  = (__bf16)a0.w;
    A[4]  = (__bf16)a1.x; A[5]  = (__bf16)a1.y; A[6]  = (__bf16)a1.z; A[7]  = (__bf16)a1.w;
    A[8]  = (__bf16)a2.x; A[9]  = (__bf16)a2.y; A[10] = (__bf16)a2.z; A[11] = (__bf16)a2.w;
    A[12] = (__bf16)a3.x; A[13] = (__bf16)a3.y; A[14] = (__bf16)a3.z; A[15] = (__bf16)a3.w;
    v16bf Bv = *(const v16bf*)(brow + k0 + 16 * h);
    acc = __builtin_amdgcn_wmma_f32_16x16x32_bf16(
        /*neg_a=*/false, A, /*neg_b=*/false, Bv,
        /*c_mod=*/(short)0, acc, /*reuse_a=*/false, /*reuse_b=*/false);
  }

  int sBase = sTile * 16 + 8 * h;
  #pragma unroll
  for (int r = 0; r < 8; ++r) {
    out[((size_t)(b * S_ + sBase + r)) * DK_ + dTile * 16 + m] = acc[r];
  }
}

// ---------------------------------------------------------------------------
// Launch. d_in: x, theta_q, theta_k, W_v, b_v.
// d_out: [ out (B*S*DK f32) | attn (B*S*S f32) ].
// d_ws usage: qst 1MB | kst 1MB | vT(bf16) 1MB  == 3MB total.
// ---------------------------------------------------------------------------
extern "C" void kernel_launch(void* const* d_in, const int* in_sizes, int n_in,
                              void* d_out, int out_size, void* d_ws, size_t ws_size,
                              hipStream_t stream) {
  (void)in_sizes; (void)n_in; (void)out_size; (void)ws_size;
  const float* x   = (const float*)d_in[0];
  const float* thq = (const float*)d_in[1];
  const float* thk = (const float*)d_in[2];
  const float* Wv  = (const float*)d_in[3];
  const float* bv  = (const float*)d_in[4];

  float* out  = (float*)d_out;
  float* attn = out + (size_t)B_ * S_ * DK_;

  float4* qst = (float4*)d_ws;
  float4* kst = qst + (size_t)B_ * S_ * NQ_;                 // +1MB
  __hip_bfloat16* vT = (__hip_bfloat16*)(kst + (size_t)B_ * S_ * NQ_);  // +1MB

  qsa_states<<<(B_ * S_ * NQ_) / 256, 256, 0, stream>>>(x, thq, thk, qst, kst);
  qsa_values<<<B_ * S_, DK_, 0, stream>>>(x, Wv, bv, vT);
  qsa_attn<<<B_ * S_, 256, 0, stream>>>(qst, kst, attn);
  qsa_out_wmma<<<(B_ * S_ * (S_ / 16) * (DK_ / 16)) / (S_ * 8), 256, 0, stream>>>(attn, vT, out);
}
